// StackedGRUEncoder_28295244546211
// MI455X (gfx1250) — matrix operand order repaired and checked by reading, not compiled, not run on hardware
//
#include <hip/hip_runtime.h>
#include <hip/hip_bf16.h>
#include <math.h>

// ---------------------------------------------------------------------------
// StackedGRUEncoder for MI455X (gfx1250, wave32, WMMA bf16 16x16x32)
//  - input projections: dense WMMA GEMM, 64x64 tile per wave
//  - recurrence: one persistent kernel per layer; Wh slice staged into LDS
//    with async global->LDS copies (ASYNCcnt), grid-wide step barrier
// ---------------------------------------------------------------------------

typedef __bf16 bf16_t;
typedef __attribute__((ext_vector_type(16))) __bf16 v16bf;
typedef __attribute__((ext_vector_type(8)))  __bf16 v8bf;
typedef __attribute__((ext_vector_type(8)))  float  v8f;

#define WMMA_BF16(a, b, c) \
  __builtin_amdgcn_wmma_f32_16x16x32_bf16(false, (a), false, (b), (short)0, (c), false, false)

// Per-lane async 16B copy: global (VGPR pair address, GV mode) -> LDS (VGPR
// byte address). Tracked by ASYNCcnt; no destination VGPRs.
static __device__ __forceinline__ void async_copy_b128(unsigned lds_addr,
                                                       const void* gaddr) {
  asm volatile("global_load_async_to_lds_b128 %0, %1, off"
               :
               : "v"(lds_addr), "v"((unsigned long long)(uintptr_t)gaddr)
               : "memory");
}
static __device__ __forceinline__ void wait_asynccnt0() {
  asm volatile("s_wait_asynccnt 0x0" ::: "memory");
}

// A fragment: 16x32 bf16 tile of row-major A (lda elems/row), tile origin (m0,k0).
// Layout (contiguous K, same as B fragments): lanes 0-15 hold row m, K = k0..k0+15;
//             lanes 16-31 hold row m, K = k0+16..k0+31.
static __device__ __forceinline__ v16bf load_frag_a(const bf16_t* A,
                                                    int lda, int m0, int k0, int lane) {
  const int m = m0 + (lane & 15);
  const bf16_t* p = A + (size_t)m * lda + k0;
  const int off_lo = (lane & 16) ? 16 : 0;
  const int off_hi = (lane & 16) ? 24 : 8;
  v8bf lo = *(const v8bf*)(p + off_lo);
  v8bf hi = *(const v8bf*)(p + off_hi);
  v16bf r;
#pragma unroll
  for (int i = 0; i < 8; ++i) { r[i] = lo[i]; r[i + 8] = hi[i]; }
  return r;
}

// B fragment from global: 32x16 tile where B(k,n) = W[n, k], W row-major.
// lane 0-15 -> col n, K = k0..k0+15 ; lane 16-31 -> col n, K = k0+16..k0+31.
static __device__ __forceinline__ v16bf load_frag_b(const bf16_t* W,
                                                    int ldw, int n0, int k0, int lane) {
  const int n = n0 + (lane & 15);
  const int k = k0 + ((lane & 16) ? 16 : 0);
  return *(const v16bf*)(W + (size_t)n * ldw + k);
}

// B fragment from LDS (padded pitch, 16B-aligned halves -> 2x ds_load_b128).
static __device__ __forceinline__ v16bf lds_frag_b(const bf16_t* sW,
                                                   int pitch, int row0, int k0, int lane) {
  const int c = row0 + (lane & 15);
  const int k = k0 + ((lane & 16) ? 16 : 0);
  const bf16_t* p = sW + (size_t)c * pitch + k;
  v8bf lo = *(const v8bf*)p;
  v8bf hi = *(const v8bf*)(p + 8);
  v16bf r;
#pragma unroll
  for (int i = 0; i < 8; ++i) { r[i] = lo[i]; r[i + 8] = hi[i]; }
  return r;
}

// ---------------------------------------------------------------------------
// C[M,N] = A[M,K](bf16) * W[N,K]^T(bf16) + bias[N], f32 out.
// One wave computes a 64x64 tile: 16 f32 accumulators, 16 WMMAs per K-step.
// ---------------------------------------------------------------------------
__global__ __launch_bounds__(256) void gemm_xt_wt_bias(
    const bf16_t* __restrict__ X, const bf16_t* __restrict__ W,
    const float* __restrict__ bias, float* __restrict__ C,
    int M, int N, int K)
{
  const int lane   = threadIdx.x & 31;
  const int waveId = (blockIdx.x << 3) | (threadIdx.x >> 5);
  const int nBlkN  = N >> 6;
  const int bm     = waveId / nBlkN;
  const int bn     = waveId - bm * nBlkN;
  if (bm >= (M >> 6)) return;                    // wave-uniform guard (EXEC stays full)
  const int m0 = bm << 6, n0 = bn << 6;

  v8f acc[4][4] = {};
  for (int k0 = 0; k0 < K; k0 += 32) {
    v16bf a[4], b[4];
#pragma unroll
    for (int i = 0; i < 4; ++i) a[i] = load_frag_a(X, K, m0 + 16 * i, k0, lane);
#pragma unroll
    for (int j = 0; j < 4; ++j) b[j] = load_frag_b(W, K, n0 + 16 * j, k0, lane);
#pragma unroll
    for (int i = 0; i < 4; ++i)
#pragma unroll
      for (int j = 0; j < 4; ++j) acc[i][j] = WMMA_BF16(a[i], b[j], acc[i][j]);
  }

  const int nl = lane & 15;
  const int mb = (lane & 16) ? 8 : 0;
#pragma unroll
  for (int i = 0; i < 4; ++i) {
#pragma unroll
    for (int j = 0; j < 4; ++j) {
      const int n = n0 + 16 * j + nl;
      const float bv = bias[n];
      float* cp = C + (size_t)(m0 + 16 * i + mb) * N + n;
#pragma unroll
      for (int r = 0; r < 8; ++r) cp[(size_t)r * N] = acc[i][j][r] + bv;
    }
  }
}

// ---------------------------------------------------------------------------
// Persistent per-layer recurrence. Grid: 64 blocks x 128 threads (4 waves).
// Block -> 16-col strip of H (stages its 48x1024 Wh slice, ~97KB, in LDS once
// via async global->LDS copies). Wave -> 16 batch rows. Per step: 3 gate WMMA
// accumulations (B from LDS), fused gate math, h ping-pong, device-wide
// step barrier.
// mode 0: also write h_new (bf16) to next layer's input at reversed time slot.
// mode 1: also write h_new (f32, masked) to final output [b, L-1-t, n].
// ---------------------------------------------------------------------------
#define WH_PITCH 1032   // 1024 + 8 bf16 -> 516 dwords -> LDS bank stride 4

__global__ __launch_bounds__(128) void gru_layer_persistent(
    const float*  __restrict__ gx,        // (L, B, 3H)
    const bf16_t* __restrict__ Whb,       // (3H, H) bf16
    const float*  __restrict__ bh,        // (3H)
    float*  h_f0, float*  h_f1,           // (B, H) f32 ping-pong
    bf16_t* h_b0, bf16_t* h_b1,           // (B, H) bf16 ping-pong
    bf16_t* __restrict__ x_next,          // (L*B, H) bf16 (mode 0)
    float*  __restrict__ out_final,       // (B, L, H) f32 (mode 1)
    const float* __restrict__ mask,       // (B, L)
    unsigned* bar, int L_, int mode)
{
  const int H_ = 1024, B_ = 64, G_ = 3 * 1024;
  extern __shared__ char smem[];
  bf16_t* sWh = (bf16_t*)smem;            // [48][WH_PITCH]

  const int n0   = blockIdx.x << 4;       // column strip of H
  const int lane = threadIdx.x & 31;
  const int wave = threadIdx.x >> 5;
  const int m0   = wave << 4;             // batch-row strip

  // ---- Stage Wh rows {g*H + n0 + c : g<3, c<16} into LDS via ASYNC DMA ----
  {
    const unsigned lds_base = (unsigned)(uintptr_t)sWh;  // flat->LDS: addr[31:0]
    for (int idx = threadIdx.x; idx < 48 * 128; idx += blockDim.x) {
      const int row = idx >> 7;           // 0..47  (g*16 + c)
      const int kc  = idx & 127;          // 8-element (16B) chunk
      const int g   = row >> 4, c = row & 15;
      const bf16_t* src = Whb + (size_t)(g * H_ + n0 + c) * H_ + kc * 8;
      const unsigned dst = lds_base + (unsigned)(row * WH_PITCH + kc * 8) * 2u;
      async_copy_b128(dst, src);
    }
    wait_asynccnt0();
  }
  __syncthreads();

  const int nl = lane & 15;
  const int mb = (lane & 16) ? 8 : 0;
  const int n  = n0 + nl;
  const float bhr = bh[n], bhz = bh[H_ + n], bhn = bh[2 * H_ + n];

  for (int t = 0; t < L_; ++t) {
    const float*  gx_t   = gx + (size_t)t * B_ * G_;
    const float*  hin_f  = (t & 1) ? h_f1 : h_f0;
    const bf16_t* hin_b  = (t & 1) ? h_b1 : h_b0;
    float*        hout_f = (t & 1) ? h_f0 : h_f1;
    bf16_t*       hout_b = (t & 1) ? h_b0 : h_b1;

    v8f accR = {}, accZ = {}, accN = {};
    for (int k0 = 0; k0 < H_; k0 += 32) {
      v16bf a  = load_frag_a(hin_b, H_, m0, k0, lane);
      v16bf br = lds_frag_b(sWh, WH_PITCH, 0,  k0, lane);
      v16bf bz = lds_frag_b(sWh, WH_PITCH, 16, k0, lane);
      v16bf bn = lds_frag_b(sWh, WH_PITCH, 32, k0, lane);
      accR = WMMA_BF16(a, br, accR);
      accZ = WMMA_BF16(a, bz, accZ);
      accN = WMMA_BF16(a, bn, accN);
    }

    const int lo = L_ - 1 - t;
#pragma unroll
    for (int r = 0; r < 8; ++r) {
      const int b = m0 + mb + r;
      const size_t gxrow = (size_t)b * G_;
      const float ghr = accR[r] + bhr;
      const float ghz = accZ[r] + bhz;
      const float ghn = accN[r] + bhn;
      const float xr = gx_t[gxrow + n];
      const float xz = gx_t[gxrow + H_ + n];
      const float xn = gx_t[gxrow + 2 * H_ + n];
      const float rg = 1.0f / (1.0f + __expf(-(xr + ghr)));
      const float zg = 1.0f / (1.0f + __expf(-(xz + ghz)));
      const float ng = tanhf(xn + rg * ghn);
      const float hp = hin_f[(size_t)b * H_ + n];
      const float hv = (1.0f - zg) * ng + zg * hp;
      hout_f[(size_t)b * H_ + n] = hv;
      hout_b[(size_t)b * H_ + n] = (bf16_t)hv;
      if (mode == 0) {
        x_next[((size_t)lo * B_ + b) * H_ + n] = (bf16_t)hv;
      } else {
        out_final[((size_t)b * L_ + lo) * H_ + n] = hv * mask[(size_t)b * L_ + lo];
      }
    }

    // Prefetch next timestep's gx columns for this thread (global_prefetch_b8).
    if (t + 1 < L_) {
      const float* nx = gx + (size_t)(t + 1) * B_ * G_ + (size_t)(m0 + mb) * G_;
      __builtin_prefetch(nx + n, 0, 0);
      __builtin_prefetch(nx + H_ + n, 0, 0);
      __builtin_prefetch(nx + 2 * H_ + n, 0, 0);
    }

    // ---- Device-wide step barrier (release h writes, acquire for next read) ----
    __threadfence();
    __syncthreads();
    if (threadIdx.x == 0) {
      __hip_atomic_fetch_add(bar, 1u, __ATOMIC_RELEASE, __HIP_MEMORY_SCOPE_AGENT);
      const unsigned target = (unsigned)(t + 1) * gridDim.x;
      while (__hip_atomic_load(bar, __ATOMIC_ACQUIRE, __HIP_MEMORY_SCOPE_AGENT) < target) {
        __builtin_amdgcn_s_sleep(2);
      }
    }
    __syncthreads();
  }
}

// ---------------------------------------------------------------------------
// Small helper kernels
// ---------------------------------------------------------------------------
__global__ void cvt_f32_bf16(const float* __restrict__ s, bf16_t* __restrict__ d, size_t n) {
  size_t i = (size_t)blockIdx.x * blockDim.x + threadIdx.x;
  if (i < n) d[i] = (bf16_t)s[i];
}

// xs (B, L, E) f32 -> X (L, B, E) bf16 (time-major for the gx GEMM)
__global__ void transpose_bl_to_lb_bf16(const float* __restrict__ xs, bf16_t* __restrict__ X,
                                        int B_, int L_, int E_) {
  size_t i = (size_t)blockIdx.x * blockDim.x + threadIdx.x;
  size_t total = (size_t)B_ * L_ * E_;
  if (i >= total) return;
  int e = (int)(i % E_);
  size_t q = i / E_;
  int b = (int)(q % B_);
  int l = (int)(q / B_);
  X[i] = (bf16_t)xs[((size_t)b * L_ + l) * E_ + e];
}

__global__ void init_state(float* __restrict__ hf, bf16_t* __restrict__ hb,
                           unsigned* __restrict__ bar, int n) {
  int i = blockIdx.x * blockDim.x + threadIdx.x;
  if (i < n) { hf[i] = 0.0f; hb[i] = (bf16_t)0.0f; }
  if (i == 0) *bar = 0u;
}

// ---------------------------------------------------------------------------
extern "C" void kernel_launch(void* const* d_in, const int* in_sizes, int n_in,
                              void* d_out, int out_size, void* d_ws, size_t ws_size,
                              hipStream_t stream)
{
  (void)in_sizes; (void)n_in; (void)out_size; (void)ws_size;
  const int B_ = 64, L_ = 512, E_ = 1024, H_ = 1024, G_ = 3 * H_;

  const float* xs   = (const float*)d_in[0];
  const float* mask = (const float*)d_in[1];
  const float* Wi0  = (const float*)d_in[2];
  const float* Wh0  = (const float*)d_in[3];
  const float* bi0  = (const float*)d_in[4];
  const float* bh0  = (const float*)d_in[5];
  const float* Wi1  = (const float*)d_in[6];
  const float* Wh1  = (const float*)d_in[7];
  const float* bi1  = (const float*)d_in[8];
  const float* bh1  = (const float*)d_in[9];
  float* out = (float*)d_out;

  // Workspace carve-up (all 256B aligned).
  char* wsp = (char*)d_ws;
  size_t off = 0;
  auto carve = [&](size_t bytes) -> void* {
    void* p = wsp + off;
    off += (bytes + 255) & ~(size_t)255;
    return p;
  };
  bf16_t* wWi0 = (bf16_t*)carve((size_t)G_ * E_ * 2);
  bf16_t* wWh0 = (bf16_t*)carve((size_t)G_ * H_ * 2);
  bf16_t* wWi1 = (bf16_t*)carve((size_t)G_ * H_ * 2);
  bf16_t* wWh1 = (bf16_t*)carve((size_t)G_ * H_ * 2);
  bf16_t* Xseq = (bf16_t*)carve((size_t)L_ * B_ * E_ * 2);  // layer input, time-major
  float*  gx   = (float*)carve((size_t)L_ * B_ * G_ * 4);   // input projections (f32)
  float*  h_f0 = (float*)carve((size_t)B_ * H_ * 4);
  float*  h_f1 = (float*)carve((size_t)B_ * H_ * 4);
  bf16_t* h_b0 = (bf16_t*)carve((size_t)B_ * H_ * 2);
  bf16_t* h_b1 = (bf16_t*)carve((size_t)B_ * H_ * 2);
  unsigned* bar = (unsigned*)carve(256);

  // 1) Weights f32 -> bf16 (resident in the 192MB L2 thereafter)
  {
    size_t n = (size_t)G_ * E_;
    int blocks = (int)((n + 255) / 256);
    cvt_f32_bf16<<<blocks, 256, 0, stream>>>(Wi0, wWi0, n);
    cvt_f32_bf16<<<blocks, 256, 0, stream>>>(Wh0, wWh0, n);
    cvt_f32_bf16<<<blocks, 256, 0, stream>>>(Wi1, wWi1, n);
    cvt_f32_bf16<<<blocks, 256, 0, stream>>>(Wh1, wWh1, n);
  }

  // 2) xs (B,L,E) -> Xseq (L,B,E) bf16
  {
    size_t n = (size_t)L_ * B_ * E_;
    transpose_bl_to_lb_bf16<<<(int)((n + 255) / 256), 256, 0, stream>>>(xs, Xseq, B_, L_, E_);
  }

  const size_t smem_bytes = (size_t)48 * WH_PITCH * sizeof(bf16_t);  // ~97 KB of 320 KB/WGP

  for (int layer = 0; layer < 2; ++layer) {
    const bf16_t* Wib = layer ? wWi1 : wWi0;
    const bf16_t* Whb = layer ? wWh1 : wWh0;
    const float*  bi  = layer ? bi1 : bi0;
    const float*  bh  = layer ? bh1 : bh0;

    // 3) gx = Xseq * Wi^T + bi  (M=32768, N=3072, K=1024; 64x64 tile per wave)
    {
      const int M = L_ * B_;
      const int waves = (M >> 6) * (G_ >> 6);   // 512 * 48 = 24576
      gemm_xt_wt_bias<<<waves / 8, 256, 0, stream>>>(Xseq, Wib, bi, gx, M, G_, E_);
    }

    // 4) h0 = 0, barrier counter = 0
    init_state<<<(B_ * H_ + 255) / 256, 256, 0, stream>>>(h_f0, h_b0, bar, B_ * H_);

    // 5) Persistent recurrence: 64 blocks (one per 16-col strip), all 512 steps.
    //    Layer 0 writes the time-reversed layer-1 input back into Xseq
    //    (its previous contents were consumed by the GEMM above).
    gru_layer_persistent<<<H_ / 16, 128, smem_bytes, stream>>>(
        gx, Whb, bh, h_f0, h_f1, h_b0, h_b1,
        Xseq, out, mask, bar, L_, layer);
  }
}
